// AdapterController_55104430408043
// MI455X (gfx1250) — matrix operand order repaired and checked by reading, not compiled, not run on hardware
//
#include <hip/hip_runtime.h>
#include <hip/hip_bf16.h>

#define BB 64
#define SS 1024
#define DD 1024
#define HH 64
#define EE 8
#define EPSF 1e-5f

typedef __bf16 bf16_t;
typedef bf16_t bf16x16 __attribute__((ext_vector_type(16)));
typedef float f32x8 __attribute__((ext_vector_type(8)));
typedef unsigned int u32x4 __attribute__((ext_vector_type(4)));

union ABf { bf16x16 v; u32x4 q[2]; };

__device__ __forceinline__ unsigned short f32_to_bf16(float f) {
    unsigned int u = __builtin_bit_cast(unsigned int, f);
    u += 0x7FFFu + ((u >> 16) & 1u);   // round to nearest even
    return (unsigned short)(u >> 16);
}
__device__ __forceinline__ float bf16_to_f32(unsigned short h) {
    unsigned int u = ((unsigned int)h) << 16;
    return __builtin_bit_cast(float, u);
}

// ---------------------------------------------------------------------------
// Kernel 0: convert + swizzle expert weights fp32 -> bf16 in WMMA B-fragment
// order. Element layout per fragment: ((fragid*2 + h)*32 + lane)*8 + jj,
// fragid = nt*KT + kt;  k = kt*32 + (lane>>4)*16 + h*8 + jj;  n = nt*16 + (lane&15)
// so the GEMM kernel loads each half-fragment with one fully-coalesced b128.
// ---------------------------------------------------------------------------
__global__ __launch_bounds__(256) void prep_weights_kernel(
    const float* __restrict__ Wdn, const float* __restrict__ Wup,
    unsigned short* __restrict__ wdsw, unsigned short* __restrict__ wusw)
{
    int t = blockIdx.x * 256 + threadIdx.x;           // 0 .. 2*E*D*H-1
    const int per = EE * DD * HH;                     // 524288
    bool isUp = t >= per;
    int o  = isUp ? (t - per) : t;
    int e  = o >> 16;                                 // D*H = 65536 per expert
    int oo = o & 65535;

    int jj   = oo & 7;
    int l    = (oo >> 3) & 31;
    int h    = (oo >> 8) & 1;
    int frag = oo >> 9;
    int KT   = isUp ? (HH / 32) : (DD / 32);          // 2 or 32
    int kt   = frag % KT;
    int nt   = frag / KT;
    int khi  = l >> 4;
    int k    = kt * 32 + khi * 16 + h * 8 + jj;
    int n    = nt * 16 + (l & 15);

    if (isUp) {  // Wu: [E, H(k), D(n)]
        wusw[(size_t)e * 65536 + oo] = f32_to_bf16(Wup[((size_t)e * HH + k) * DD + n]);
    } else {     // Wd: [E, D(k), H(n)]
        wdsw[(size_t)e * 65536 + oo] = f32_to_bf16(Wdn[((size_t)e * DD + k) * HH + n]);
    }
}

// ---------------------------------------------------------------------------
// Kernel 1: LayerNorm per token, write z as bf16 row-major [B*S, D]
// ---------------------------------------------------------------------------
__global__ __launch_bounds__(256) void layernorm_kernel(
    const float* __restrict__ x, const float* __restrict__ ln_g,
    const float* __restrict__ ln_b, unsigned short* __restrict__ zbf)
{
    const int tok = blockIdx.x;               // 0 .. B*S-1
    const int tid = threadIdx.x;
    const float4 v = ((const float4*)(x + (size_t)tok * DD))[tid];
    float s = v.x + v.y + v.z + v.w;
    float q = v.x * v.x + v.y * v.y + v.z * v.z + v.w * v.w;
    #pragma unroll
    for (int off = 16; off > 0; off >>= 1) {
        s += __shfl_xor(s, off, 32);
        q += __shfl_xor(q, off, 32);
    }
    __shared__ float2 red[8];
    if ((tid & 31) == 0) red[tid >> 5] = make_float2(s, q);
    __syncthreads();
    s = 0.f; q = 0.f;
    #pragma unroll
    for (int i = 0; i < 8; ++i) { s += red[i].x; q += red[i].y; }
    const float mean = s * (1.0f / DD);
    const float var  = q * (1.0f / DD) - mean * mean;
    const float rs   = rsqrtf(var + EPSF);

    const float4 gg = ((const float4*)ln_g)[tid];
    const float4 bb = ((const float4*)ln_b)[tid];
    ushort4 outv;
    outv.x = f32_to_bf16((v.x - mean) * rs * gg.x + bb.x);
    outv.y = f32_to_bf16((v.y - mean) * rs * gg.y + bb.y);
    outv.z = f32_to_bf16((v.z - mean) * rs * gg.z + bb.z);
    outv.w = f32_to_bf16((v.w - mean) * rs * gg.w + bb.w);
    ((ushort4*)(zbf + (size_t)tok * DD))[tid] = outv;
}

// ---------------------------------------------------------------------------
// Kernel 2: sentence embedding  z_sent[b,d] = mean_s z[b,s,d]
// ---------------------------------------------------------------------------
__global__ __launch_bounds__(256) void sentmean_kernel(
    const unsigned short* __restrict__ zbf, float* __restrict__ zs)
{
    int t = blockIdx.x * 256 + threadIdx.x;   // 0 .. B*D-1
    int b = t >> 10;
    int d = t & (DD - 1);
    const unsigned short* p = zbf + (size_t)b * SS * DD + d;
    float s = 0.f;
    for (int i = 0; i < SS; ++i) s += bf16_to_f32(p[(size_t)i * DD]);
    zs[t] = s * (1.0f / SS);
}

// ---------------------------------------------------------------------------
// Kernel 3: BatchNorm(eval) -> router linear D->8 -> softmax -> top-1
// ---------------------------------------------------------------------------
__global__ __launch_bounds__(256) void router_kernel(
    const float* __restrict__ zs, const float* __restrict__ bng,
    const float* __restrict__ bnb, const float* __restrict__ bnm,
    const float* __restrict__ bnv, const float* __restrict__ Wr,
    const float* __restrict__ br, int* __restrict__ idxo, float* __restrict__ gateo)
{
    const int b = blockIdx.x, tid = threadIdx.x;
    float p[EE];
    #pragma unroll
    for (int e = 0; e < EE; ++e) p[e] = 0.f;
    for (int d = tid; d < DD; d += 256) {
        float zb = (zs[b * DD + d] - bnm[d]) * rsqrtf(bnv[d] + EPSF) * bng[d] + bnb[d];
        const float* wr = Wr + d * EE;
        #pragma unroll
        for (int e = 0; e < EE; ++e) p[e] += zb * wr[e];
    }
    #pragma unroll
    for (int e = 0; e < EE; ++e)
        #pragma unroll
        for (int off = 16; off > 0; off >>= 1) p[e] += __shfl_xor(p[e], off, 32);
    __shared__ float red[8][EE];
    if ((tid & 31) == 0) {
        #pragma unroll
        for (int e = 0; e < EE; ++e) red[tid >> 5][e] = p[e];
    }
    __syncthreads();
    if (tid == 0) {
        float lg[EE];
        #pragma unroll
        for (int e = 0; e < EE; ++e) {
            float s = br[e];
            #pragma unroll
            for (int w = 0; w < 8; ++w) s += red[w][e];
            lg[e] = s;
        }
        float mx = lg[0]; int mi = 0;
        #pragma unroll
        for (int e = 1; e < EE; ++e) if (lg[e] > mx) { mx = lg[e]; mi = e; }
        float sum = 0.f;
        #pragma unroll
        for (int e = 0; e < EE; ++e) sum += __expf(lg[e] - mx);
        idxo[b]  = mi;
        gateo[b] = 1.0f / sum;    // softmax prob of the argmax expert
    }
}

// ---------------------------------------------------------------------------
// Kernel 4: fused bottleneck adapter per (batch, 64-token tile)
//   h = relu(z @ Wd[e] + bd[e]);  y = (h @ Wu[e] + bu[e]) * gate
// 8 waves: wave = (m-tile 0..3) x (half 0..1). WMMA bf16 16x16x32, f32 accum.
// Wu[e] (128 KB, pre-swizzled) is staged into LDS with async global->LDS
// copies issued up-front, completing in the shadow of GEMM1.
// ---------------------------------------------------------------------------
__global__ __launch_bounds__(256) void moe_adapter_kernel(
    const unsigned short* __restrict__ zbf,
    const unsigned short* __restrict__ wdsw,
    const unsigned short* __restrict__ wusw,
    const float* __restrict__ b_down, const float* __restrict__ b_up,
    const int* __restrict__ idx, const float* __restrict__ gate,
    float* __restrict__ y)
{
    const int bx    = blockIdx.x;
    const int b     = bx >> 4;           // batch
    const int tile  = bx & 15;           // 64-token tile within S
    const int tid   = threadIdx.x;
    const int w     = tid >> 5;          // wave 0..7
    const int l     = tid & 31;
    const int l16   = l & 15;
    const int khi   = l >> 4;
    const int m     = w >> 1;            // M-tile 0..3
    const int hpair = w & 1;

    const int   e  = __builtin_amdgcn_readfirstlane(idx[b]);
    const float g  = gate[b];

    __shared__ __align__(16) unsigned short wush[HH * DD];  // staged Wu[e], 128 KB
    __shared__ __align__(16) unsigned short hsh[64 * 72];   // h tile, bf16, padded rows

    // ---- issue async global->LDS staging of the whole pre-swizzled Wu[e] ----
    {
        const unsigned long long wubase =
            (unsigned long long)(uintptr_t)(wusw + (size_t)e * (HH * DD));
        const unsigned ldsbase = (unsigned)(uintptr_t)(&wush[0]);
        #pragma unroll
        for (int i = 0; i < 32; ++i) {
            unsigned off = (unsigned)tid * 16u + (unsigned)i * 4096u;  // bytes
            asm volatile("global_load_async_to_lds_b128 %0, %1, %2"
                         :: "v"(ldsbase + off), "v"(off), "s"(wubase)
                         : "memory");
        }
    }

    // ---------------- GEMM1: [64 x 1024] x [1024 x 64] -> h ----------------
    const size_t zrow = ((size_t)b * SS + (size_t)tile * 64 + m * 16 + l16) * DD;
    const unsigned short* wd = wdsw + (size_t)e * DD * HH;
    const int n0 = hpair * 2, n1 = n0 + 1;

    f32x8 c0 = {}, c1 = {};
    #pragma unroll 4
    for (int kt = 0; kt < DD / 32; ++kt) {
        ABf a, bf0, bf1;
        const unsigned short* za = zbf + zrow + kt * 32 + khi * 16;
        a.q[0] = *(const u32x4*)(za);
        a.q[1] = *(const u32x4*)(za + 8);
        const unsigned short* pb0 = wd + ((size_t)(n0 * 32 + kt)) * 512 + l * 8;
        const unsigned short* pb1 = wd + ((size_t)(n1 * 32 + kt)) * 512 + l * 8;
        bf0.q[0] = *(const u32x4*)(pb0);
        bf0.q[1] = *(const u32x4*)(pb0 + 256);
        bf1.q[0] = *(const u32x4*)(pb1);
        bf1.q[1] = *(const u32x4*)(pb1 + 256);
        c0 = __builtin_amdgcn_wmma_f32_16x16x32_bf16(false, a.v, false, bf0.v,
                                                     (short)0, c0, false, false);
        c1 = __builtin_amdgcn_wmma_f32_16x16x32_bf16(false, a.v, false, bf1.v,
                                                     (short)0, c1, false, false);
    }

    // bias + ReLU, stash h into LDS as bf16
    const float bd0 = b_down[e * HH + n0 * 16 + l16];
    const float bd1 = b_down[e * HH + n1 * 16 + l16];
    #pragma unroll
    for (int r = 0; r < 8; ++r) {
        const int row = m * 16 + r + khi * 8;
        hsh[row * 72 + n0 * 16 + l16] = f32_to_bf16(fmaxf(c0[r] + bd0, 0.0f));
        hsh[row * 72 + n1 * 16 + l16] = f32_to_bf16(fmaxf(c1[r] + bd1, 0.0f));
    }
    // wait for this wave's async Wu staging, then block-wide barrier
    asm volatile("s_wait_asynccnt 0x0" ::: "memory");
    __syncthreads();

    // ---------------- GEMM2: [64 x 64] x [64 x 1024] -> y ------------------
    ABf a0, a1;
    {
        const unsigned short* pa = &hsh[(m * 16 + l16) * 72];
        a0.q[0] = *(const u32x4*)(pa + khi * 16);
        a0.q[1] = *(const u32x4*)(pa + khi * 16 + 8);
        a1.q[0] = *(const u32x4*)(pa + 32 + khi * 16);
        a1.q[1] = *(const u32x4*)(pa + 32 + khi * 16 + 8);
    }
    float* yb = y + (size_t)b * SS * DD;
    const int growb = tile * 64 + m * 16;

    #pragma unroll 2
    for (int i = 0; i < 32; ++i) {
        const int nt = 2 * i + hpair;
        const unsigned short* pb = wush + (size_t)nt * 1024 + l * 8;  // LDS
        ABf bf0, bf1;
        bf0.q[0] = *(const u32x4*)(pb);          // kt=0, half 0
        bf0.q[1] = *(const u32x4*)(pb + 256);    // kt=0, half 1
        bf1.q[0] = *(const u32x4*)(pb + 512);    // kt=1, half 0
        bf1.q[1] = *(const u32x4*)(pb + 768);    // kt=1, half 1
        f32x8 c = {};
        c = __builtin_amdgcn_wmma_f32_16x16x32_bf16(false, a0.v, false, bf0.v,
                                                    (short)0, c, false, false);
        c = __builtin_amdgcn_wmma_f32_16x16x32_bf16(false, a1.v, false, bf1.v,
                                                    (short)0, c, false, false);
        const int   col = nt * 16 + l16;
        const float bu  = b_up[e * DD + col];
        #pragma unroll
        for (int r = 0; r < 8; ++r) {
            const int row = growb + r + khi * 8;
            yb[(size_t)row * DD + col] = (c[r] + bu) * g;
        }
    }
}

// ---------------------------------------------------------------------------
extern "C" void kernel_launch(void* const* d_in, const int* in_sizes, int n_in,
                              void* d_out, int out_size, void* d_ws, size_t ws_size,
                              hipStream_t stream) {
    const float* x       = (const float*)d_in[0];
    const float* ln_g    = (const float*)d_in[1];
    const float* ln_b    = (const float*)d_in[2];
    const float* bn_g    = (const float*)d_in[3];
    const float* bn_b    = (const float*)d_in[4];
    const float* bn_mean = (const float*)d_in[5];
    const float* bn_var  = (const float*)d_in[6];
    const float* Wr      = (const float*)d_in[7];
    const float* br      = (const float*)d_in[8];
    const float* W_down  = (const float*)d_in[9];
    const float* b_down  = (const float*)d_in[10];
    const float* W_up    = (const float*)d_in[11];
    const float* b_up    = (const float*)d_in[12];
    float* out           = (float*)d_out;

    char* ws = (char*)d_ws;
    size_t off = 0;
    unsigned short* zbf = (unsigned short*)(ws + off); off += (size_t)BB * SS * DD * 2;
    float* zs           = (float*)(ws + off);          off += (size_t)BB * DD * 4;
    float* gate         = (float*)(ws + off);          off += 256;
    int*   eidx         = (int*)(ws + off);            off += 256;
    unsigned short* wdsw = (unsigned short*)(ws + off); off += (size_t)EE * DD * HH * 2;
    unsigned short* wusw = (unsigned short*)(ws + off); off += (size_t)EE * HH * DD * 2;

    prep_weights_kernel<<<(2 * EE * DD * HH) / 256, 256, 0, stream>>>(W_down, W_up, wdsw, wusw);
    layernorm_kernel<<<BB * SS, 256, 0, stream>>>(x, ln_g, ln_b, zbf);
    sentmean_kernel<<<(BB * DD) / 256, 256, 0, stream>>>(zbf, zs);
    router_kernel<<<BB, 256, 0, stream>>>(zs, bn_g, bn_b, bn_mean, bn_var, Wr, br, eidx, gate);
    moe_adapter_kernel<<<BB * (SS / 64), 256, 0, stream>>>(zbf, wdsw, wusw, b_down, b_up,
                                                           eidx, gate, out);
}
